// STMGNN_84224308674631
// MI455X (gfx1250) — compile-verified, matrix operand order checked by reading, not run on hardware
//
#include <hip/hip_runtime.h>
#include <math.h>

typedef __attribute__((ext_vector_type(16))) _Float16 v16h;
typedef __attribute__((ext_vector_type(8)))  float    v8f;

#define DF 64
#define HH 8
#define CC 8

// ---- order-preserving float<->uint transform for atomic max on floats ----
__device__ __forceinline__ unsigned f2ord(float f) {
  unsigned u = __float_as_uint(f);
  return (u & 0x80000000u) ? ~u : (u | 0x80000000u);
}
__device__ __forceinline__ float ord2f(unsigned u) {
  return (u & 0x80000000u) ? __uint_as_float(u & 0x7fffffffu)
                           : __uint_as_float(~u);
}
#define ORD_NEG_INF 0x007FFFFFu   // f2ord(-inf)

// =====================================================================
// K1: Out[N,64] = act(A[N,64] @ W[64,64] + bias) via WMMA f16->f32
// One wave computes a 16x64 tile: 4 col-tiles x 2 K-steps of 16x16x32.
// W is pre-swizzled into LDS in the exact per-lane WMMA B layout, so each
// B operand is a single 32B contiguous per-lane read (2x ds_load_b128).
// =====================================================================
__global__ __launch_bounds__(256) void gat_gemm64_wmma(
    const float* __restrict__ A, const float* __restrict__ W,
    const float* __restrict__ bias, float* __restrict__ Out,
    int Nrows, int relu)
{
  // Bsw[kt(2)][nt(4)][lane(32)][i(16)] halves = 4096 halves = 8KB
  __shared__ alignas(32) _Float16 Bsw[4096];
  __shared__ float bs[DF];
  {
    const int t    = threadIdx.x;       // 256 threads = 8 tiles x 32 lanes
    const int tile = t >> 5;            // 0..7 = kt*4+nt
    const int ln   = t & 31;
    const int kt   = tile >> 2;
    const int nt   = tile & 3;
    const int kb   = (ln >> 4) * 8;
    const int n    = nt * 16 + (ln & 15);
    const int koff = kt * 32;
    v16h wv;
#pragma unroll
    for (int i = 0; i < 8; ++i) {
      wv[i]     = (_Float16)W[(koff + kb + i) * DF + n];
      wv[i + 8] = (_Float16)W[(koff + 16 + kb + i) * DF + n];
    }
    *(v16h*)&Bsw[tile * 512 + ln * 16] = wv;   // 2x ds_store_b128
  }
  if (threadIdx.x < DF) bs[threadIdx.x] = bias[threadIdx.x];
  __syncthreads();

  const int lane = threadIdx.x & 31;
  const int wave = threadIdx.x >> 5;
  const int rowBase = blockIdx.x * 128 + wave * 16;
  if (rowBase >= Nrows) return;

  const int mcol = lane & 15;          // M for A, N for B/C
  const int kb   = (lane >> 4) * 8;    // per-half-wave K offset (and M offset for C)

  v8f acc[4];
#pragma unroll
  for (int nt = 0; nt < 4; ++nt) {
    float bv = bs[nt * 16 + mcol];
#pragma unroll
    for (int r = 0; r < 8; ++r) acc[nt][r] = bv;
  }

  int rowA = rowBase + mcol;
  if (rowA >= Nrows) rowA = Nrows - 1;      // keep EXEC all-ones for WMMA
  const float4* arow4 = (const float4*)(A + (size_t)rowA * DF);

#pragma unroll
  for (int kt = 0; kt < 2; ++kt) {
    const int koff = kt * 32;
    // A tile 16x32 f16, ISA layout: elems 0..7 -> K=koff+kb+i, 8..15 -> K=koff+16+kb+i
    const float4 fa = arow4[(koff + kb) >> 2];
    const float4 fb = arow4[((koff + kb) >> 2) + 1];
    const float4 fc = arow4[(koff + 16 + kb) >> 2];
    const float4 fd = arow4[((koff + 16 + kb) >> 2) + 1];
    v16h av;
    av[0]  = (_Float16)fa.x; av[1]  = (_Float16)fa.y;
    av[2]  = (_Float16)fa.z; av[3]  = (_Float16)fa.w;
    av[4]  = (_Float16)fb.x; av[5]  = (_Float16)fb.y;
    av[6]  = (_Float16)fb.z; av[7]  = (_Float16)fb.w;
    av[8]  = (_Float16)fc.x; av[9]  = (_Float16)fc.y;
    av[10] = (_Float16)fc.z; av[11] = (_Float16)fc.w;
    av[12] = (_Float16)fd.x; av[13] = (_Float16)fd.y;
    av[14] = (_Float16)fd.z; av[15] = (_Float16)fd.w;
#pragma unroll
    for (int nt = 0; nt < 4; ++nt) {
      const v16h bv = *(const v16h*)&Bsw[(kt * 4 + nt) * 512 + lane * 16];
      acc[nt] = __builtin_amdgcn_wmma_f32_16x16x32_f16(
          false, av, false, bv, (short)0, acc[nt], false, false);
    }
  }

#pragma unroll
  for (int nt = 0; nt < 4; ++nt) {
    const int col = nt * 16 + mcol;
#pragma unroll
    for (int r = 0; r < 8; ++r) {
      int row = rowBase + r + kb;   // C layout: lanes>=16 hold M=8..15
      if (row < Nrows) {
        float v = acc[nt][r];
        if (relu) v = fmaxf(v, 0.f);
        Out[(size_t)row * DF + col] = v;
      }
    }
  }
}

// =====================================================================
// K2: per-node attention logits: adst[n,h] = hp[n,h,:].att[h,:C]
//                                asrc[n,h] = hp[n,h,:].att[h,C:]
// =====================================================================
__global__ void gat_attdot(const float* __restrict__ hp,
                           const float* __restrict__ att,   // [H, 2C]
                           float* __restrict__ adst, float* __restrict__ asrc,
                           int Nn)
{
  int i = blockIdx.x * blockDim.x + threadIdx.x;
  if (i >= Nn * HH) return;
  int n = i >> 3, h = i & 7;
  const float4* row = (const float4*)(hp + (size_t)n * DF + h * CC);
  const float4* a   = (const float4*)(att + h * (2 * CC));
  float4 r0 = row[0], r1 = row[1];
  float4 ad0 = a[0], ad1 = a[1], as0 = a[2], as1 = a[3];
  float sd = r0.x*ad0.x + r0.y*ad0.y + r0.z*ad0.z + r0.w*ad0.w
           + r1.x*ad1.x + r1.y*ad1.y + r1.z*ad1.z + r1.w*ad1.w;
  float ss = r0.x*as0.x + r0.y*as0.y + r0.z*as0.z + r0.w*as0.w
           + r1.x*as1.x + r1.y*as1.y + r1.z*as1.z + r1.w*as1.w;
  adst[i] = sd;
  asrc[i] = ss;
}

// =====================================================================
// K3: per-layer init: m = ord(-inf), s = 0, agg = 0
// =====================================================================
__global__ void gat_init(unsigned* __restrict__ mo, float* __restrict__ s,
                         float* __restrict__ agg, int Nn)
{
  int i = blockIdx.x * blockDim.x + threadIdx.x;
  if (i < Nn * DF) agg[i] = 0.f;
  if (i < Nn * HH) { mo[i] = ORD_NEG_INF; s[i] = 0.f; }
}

// =====================================================================
// K4: segment max of leaky_relu(adst[dst]+asrc[src]) over dst
// =====================================================================
__global__ void gat_edge_max(const int* __restrict__ src, const int* __restrict__ dst,
                             const float* __restrict__ adst, const float* __restrict__ asrc,
                             unsigned* __restrict__ mo, int Ee)
{
  int e = blockIdx.x * blockDim.x + threadIdx.x;
  if (e >= Ee) return;
  int s_ = src[e], d_ = dst[e];
  const float4* ad4 = (const float4*)adst + (size_t)d_ * 2;
  const float4* as4 = (const float4*)asrc + (size_t)s_ * 2;
  float4 d0 = ad4[0], d1 = ad4[1];
  float4 s0 = as4[0], s1 = as4[1];
  float al[8] = { d0.x + s0.x, d0.y + s0.y, d0.z + s0.z, d0.w + s0.w,
                  d1.x + s1.x, d1.y + s1.y, d1.z + s1.z, d1.w + s1.w };
#pragma unroll
  for (int h = 0; h < HH; ++h) {
    float a = al[h];
    a = (a > 0.f) ? a : 0.2f * a;
    atomicMax(&mo[(size_t)d_ * HH + h], f2ord(a));
  }
}

// =====================================================================
// K5: ea = exp(alpha - m[dst]); s[dst] += ea; store ea
// =====================================================================
__global__ void gat_edge_exp(const int* __restrict__ src, const int* __restrict__ dst,
                             const float* __restrict__ adst, const float* __restrict__ asrc,
                             const unsigned* __restrict__ mo, float* __restrict__ sden,
                             float* __restrict__ ea, int Ee)
{
  int e = blockIdx.x * blockDim.x + threadIdx.x;
  if (e >= Ee) return;
  int s_ = src[e], d_ = dst[e];
  const float4* ad4 = (const float4*)adst + (size_t)d_ * 2;
  const float4* as4 = (const float4*)asrc + (size_t)s_ * 2;
  const uint4*  mo4 = (const uint4*)mo + (size_t)d_ * 2;
  float4 d0 = ad4[0], d1 = ad4[1];
  float4 s0 = as4[0], s1 = as4[1];
  uint4  m0 = mo4[0], m1 = mo4[1];
  float al[8] = { d0.x + s0.x, d0.y + s0.y, d0.z + s0.z, d0.w + s0.w,
                  d1.x + s1.x, d1.y + s1.y, d1.z + s1.z, d1.w + s1.w };
  unsigned mv[8] = { m0.x, m0.y, m0.z, m0.w, m1.x, m1.y, m1.z, m1.w };
  float v[8];
#pragma unroll
  for (int h = 0; h < HH; ++h) {
    float a = al[h];
    a = (a > 0.f) ? a : 0.2f * a;
    v[h] = expf(a - ord2f(mv[h]));
  }
  float4* ea4 = (float4*)(ea + (size_t)e * HH);
  ea4[0] = make_float4(v[0], v[1], v[2], v[3]);
  ea4[1] = make_float4(v[4], v[5], v[6], v[7]);
#pragma unroll
  for (int h = 0; h < HH; ++h)
    unsafeAtomicAdd(&sden[(size_t)d_ * HH + h], v[h]);
}

// =====================================================================
// K6: message scatter: agg[dst,h,:] += hp[src,h,:] * ea[e,h]
// one thread per (edge, head): 2x b128 loads + 8 float atomics
// =====================================================================
__global__ void gat_edge_msg(const int* __restrict__ src, const int* __restrict__ dst,
                             const float* __restrict__ hp, const float* __restrict__ ea,
                             float* __restrict__ agg, int Ee)
{
  int t = blockIdx.x * blockDim.x + threadIdx.x;
  if (t >= Ee * HH) return;
  int e = t >> 3, h = t & 7;
  int s_ = src[e], d_ = dst[e];
  float w = ea[(size_t)e * HH + h];
  const float4* xin = (const float4*)(hp + (size_t)s_ * DF + h * CC);
  float4 x0 = xin[0], x1 = xin[1];
  float* aout = agg + (size_t)d_ * DF + h * CC;
  unsafeAtomicAdd(&aout[0], x0.x * w);
  unsafeAtomicAdd(&aout[1], x0.y * w);
  unsafeAtomicAdd(&aout[2], x0.z * w);
  unsafeAtomicAdd(&aout[3], x0.w * w);
  unsafeAtomicAdd(&aout[4], x1.x * w);
  unsafeAtomicAdd(&aout[5], x1.y * w);
  unsafeAtomicAdd(&aout[6], x1.z * w);
  unsafeAtomicAdd(&aout[7], x1.w * w);
}

// =====================================================================
// K7: finalize: out = LayerNorm(elu(agg/(s+eps) + ob)) * g + b
// one wave32 per node, 2 features per lane, shuffle reductions
// =====================================================================
__global__ __launch_bounds__(256) void gat_finalize(
    const float* __restrict__ agg, const float* __restrict__ sden,
    const float* __restrict__ ob, const float* __restrict__ g,
    const float* __restrict__ b, float* __restrict__ out, int Nn)
{
  int n = blockIdx.x * 8 + (threadIdx.x >> 5);
  if (n >= Nn) return;
  int lane = threadIdx.x & 31;

  float v[2];
#pragma unroll
  for (int p = 0; p < 2; ++p) {
    int idx = lane + p * 32;
    int h = idx >> 3;
    float x = agg[(size_t)n * DF + idx] / (sden[(size_t)n * HH + h] + 1e-16f) + ob[idx];
    v[p] = (x > 0.f) ? x : (expf(x) - 1.f);   // ELU
  }

  float sum = v[0] + v[1];
  float sq  = v[0] * v[0] + v[1] * v[1];
#pragma unroll
  for (int off = 16; off > 0; off >>= 1) {
    sum += __shfl_xor(sum, off, 32);
    sq  += __shfl_xor(sq,  off, 32);
  }
  float mu  = sum * (1.f / 64.f);
  float var = sq * (1.f / 64.f) - mu * mu;
  float inv = rsqrtf(var + 1e-5f);

#pragma unroll
  for (int p = 0; p < 2; ++p) {
    int idx = lane + p * 32;
    out[(size_t)n * DF + idx] = (v[p] - mu) * inv * g[idx] + b[idx];
  }
}

// =====================================================================
// host side
// =====================================================================
static inline int cdiv(int a, int b) { return (a + b - 1) / b; }

extern "C" void kernel_launch(void* const* d_in, const int* in_sizes, int n_in,
                              void* d_out, int out_size, void* d_ws, size_t ws_size,
                              hipStream_t stream) {
  const float* x     = (const float*)d_in[0];
  const int*   edge  = (const int*)  d_in[1];
  const float* W_in  = (const float*)d_in[2];
  const float* b_in  = (const float*)d_in[3];
  const float* lin_w = (const float*)d_in[4];
  const float* lin_b = (const float*)d_in[5];
  const float* att   = (const float*)d_in[6];
  const float* obias = (const float*)d_in[7];
  const float* ln_g  = (const float*)d_in[8];
  const float* ln_b  = (const float*)d_in[9];

  const int N = in_sizes[0] / DF;
  const int E = in_sizes[1] / 2;
  const int L = 5;
  const int* src = edge;
  const int* dst = edge + E;

  // workspace carve-up (all 32B aligned since chunks are multiples of 8 floats)
  char* p = (char*)d_ws;
  float*    h    = (float*)p;             p += (size_t)N * DF * sizeof(float);
  float*    hp   = (float*)p;             p += (size_t)N * DF * sizeof(float);
  float*    agg  = (float*)p;             p += (size_t)N * DF * sizeof(float);
  float*    adst = (float*)p;             p += (size_t)N * HH * sizeof(float);
  float*    asrc = (float*)p;             p += (size_t)N * HH * sizeof(float);
  unsigned* mo   = (unsigned*)p;          p += (size_t)N * HH * sizeof(unsigned);
  float*    sden = (float*)p;             p += (size_t)N * HH * sizeof(float);
  float*    ea   = (float*)p;             p += (size_t)E * HH * sizeof(float);
  (void)ws_size; (void)n_in; (void)out_size;

  dim3 blk(256);
  dim3 gGemm(cdiv(N, 128));
  dim3 gNH(cdiv(N * HH, 256));
  dim3 gN64(cdiv(N * DF, 256));
  dim3 gE(cdiv(E, 256));
  dim3 gEH(cdiv(E * HH, 256));
  dim3 gFin(cdiv(N, 8));

  // input embedding: h = relu(x @ W_in + b_in)
  gat_gemm64_wmma<<<gGemm, blk, 0, stream>>>(x, W_in, b_in, h, N, 1);

  for (int l = 0; l < L; ++l) {
    const float* Wl  = lin_w + (size_t)l * DF * DF;
    const float* bl  = lin_b + (size_t)l * DF;
    const float* al  = att   + (size_t)l * HH * 2 * CC;
    const float* obl = obias + (size_t)l * DF;
    const float* gl  = ln_g  + (size_t)l * DF;
    const float* bbl = ln_b  + (size_t)l * DF;

    gat_gemm64_wmma<<<gGemm, blk, 0, stream>>>(h, Wl, bl, hp, N, 0);
    gat_attdot     <<<gNH,   blk, 0, stream>>>(hp, al, adst, asrc, N);
    gat_init       <<<gN64,  blk, 0, stream>>>(mo, sden, agg, N);
    gat_edge_max   <<<gE,    blk, 0, stream>>>(src, dst, adst, asrc, mo, E);
    gat_edge_exp   <<<gE,    blk, 0, stream>>>(src, dst, adst, asrc, mo, sden, ea, E);
    gat_edge_msg   <<<gEH,   blk, 0, stream>>>(src, dst, hp, ea, agg, E);

    float* hout = (l == L - 1) ? (float*)d_out : h;
    gat_finalize   <<<gFin,  blk, 0, stream>>>(agg, sden, obl, gl, bbl, hout, N);
  }
}
